// MambaVisionMixer3D_26817775796995
// MI455X (gfx1250) — matrix operand order repaired and checked by reading, compiled, not run on hardware
//
#include <hip/hip_runtime.h>
#include <hip/hip_bf16.h>

// ---------------------------------------------------------------------------
// MambaVision mixer forward for MI455X (gfx1250, wave32, WMMA).
//   B=2, L=8192, D_MODEL=512, D_INNER=512, D2=256, N_STATE=16, DT_RANK=32
// Pipeline:
//   1) xz    = H @ in_proj_w^T                    (WMMA f16, 16x64/wave)
//   2) xc/zc = silu(depthwise_conv3(x|z))         (VALU)
//   3) x_dbl = xc @ x_proj_w^T                    (WMMA)
//   4) delta = softplus(x_dbl[:,:32] @ dt_w^T + b)(WMMA + fused epilogue)
//   5) y     = selective_scan, 3-pass chunked     (32x parallelism vs serial)
//   6) out   = [y | zc] @ out_proj_w^T            (WMMA)
// ---------------------------------------------------------------------------

#define BATCH   2
#define SEQLEN  8192
#define BL      (BATCH * SEQLEN)      // 16384
#define DMODEL  512
#define DINNER  512
#define D2      256
#define NSTATE  16
#define DTRANK  32
#define XDBLW   64                    // DT_RANK + 2*NSTATE
#define NCHUNK  32                    // scan chunks per sequence
#define CLEN    (SEQLEN / NCHUNK)     // 256 steps per chunk

typedef __attribute__((ext_vector_type(16))) _Float16 v16h;
typedef __attribute__((ext_vector_type(8)))  float    v8f;

// --- WMMA fragment loaders (CDNA5 ISA 7.12.2 layouts, f32->f16 convert) ----
// A 16x32 (MxK): lane<16 holds K={0..7,16..23}, lane>=16 holds K={8..15,24..31}
__device__ __forceinline__ v16h load_afrag(const float* __restrict__ arow,
                                           int k0, int lane) {
    v16h a;
    const float* p = arow + k0 + ((lane >> 4) << 3);
#pragma unroll
    for (int e = 0; e < 8; ++e) {
        a[e]     = (_Float16)p[e];        // K = base + e
        a[e + 8] = (_Float16)p[e + 16];   // K = base + 16 + e
    }
    return a;
}
// B 32x16 (KxN), column-in-lane: lane%16 = N, lane<16 holds K={0..7,16..23}, lane>=16 holds K={8..15,24..31}
__device__ __forceinline__ v16h load_bfrag(const float* __restrict__ wrow,
                                           int k0, int lane) {
    v16h b;
    const float* p = wrow + k0 + ((lane >> 4) << 3);
#pragma unroll
    for (int e = 0; e < 8; ++e) {
        b[e]     = (_Float16)p[e];
        b[e + 8] = (_Float16)p[e + 16];
    }
    return b;
}

// ---------------------------------------------------------------------------
// Y[M,ldy] = A[M,lda] * W[N,ldw]^T ; each wave owns a 16x64 strip: one A
// fragment feeds 4 WMMAs (4x A reuse). K and NG are compile-time so the
// K-loop has a known trip count: unroll-by-2 lets the scheduler overlap
// iteration i+1 loads/converts with iteration i WMMAs without predicated
// copies or spills.
// ---------------------------------------------------------------------------
template <int K, int NG>
__global__ void gemm_f16wmma4(const float* __restrict__ A,
                              const float* __restrict__ W,
                              float* __restrict__ Y,
                              int lda, int ldw, int ldy) {
    const int wave = (int)((blockIdx.x * blockDim.x + threadIdx.x) >> 5);
    const int lane = threadIdx.x & 31;
    const int tm = wave / NG, tg = wave % NG;       // NG is a power of two
    const int row0 = tm << 4, col0 = tg << 6;

    const float* arow = A + (size_t)(row0 + (lane & 15)) * lda;
    const float* wrow[4];
#pragma unroll
    for (int j = 0; j < 4; ++j)
        wrow[j] = W + (size_t)(col0 + (j << 4) + (lane & 15)) * ldw;

    v8f acc0 = {}, acc1 = {}, acc2 = {}, acc3 = {};
#pragma unroll 2
    for (int k0 = 0; k0 < K; k0 += 32) {
        if (k0 + 32 < K) {  // gfx1250 global_prefetch_b8: next K slab
            __builtin_prefetch(arow + k0 + 32 + ((lane >> 4) << 3), 0, 3);
#pragma unroll
            for (int j = 0; j < 4; ++j)
                __builtin_prefetch(wrow[j] + k0 + 32 + ((lane >> 4) << 4), 0, 3);
        }
        v16h a  = load_afrag(arow, k0, lane);
        v16h b0 = load_bfrag(wrow[0], k0, lane);
        v16h b1 = load_bfrag(wrow[1], k0, lane);
        v16h b2 = load_bfrag(wrow[2], k0, lane);
        v16h b3 = load_bfrag(wrow[3], k0, lane);
        acc0 = __builtin_amdgcn_wmma_f32_16x16x32_f16(false, a, false, b0,
                                                      (short)0, acc0, false, false);
        acc1 = __builtin_amdgcn_wmma_f32_16x16x32_f16(false, a, false, b1,
                                                      (short)0, acc1, false, false);
        acc2 = __builtin_amdgcn_wmma_f32_16x16x32_f16(false, a, false, b2,
                                                      (short)0, acc2, false, false);
        acc3 = __builtin_amdgcn_wmma_f32_16x16x32_f16(false, a, false, b3,
                                                      (short)0, acc3, false, false);
    }
    // C/D layout: lane%16 = N, rows M = i + 8*(lane/16)
    const int mb = row0 + ((lane >> 4) << 3);
    float* yr = Y + (size_t)mb * ldy + col0 + (lane & 15);
#pragma unroll
    for (int i = 0; i < 8; ++i) {
        yr[(size_t)i * ldy +  0] = acc0[i];
        yr[(size_t)i * ldy + 16] = acc1[i];
        yr[(size_t)i * ldy + 32] = acc2[i];
        yr[(size_t)i * ldy + 48] = acc3[i];
    }
}

// ---------------------------------------------------------------------------
// delta[m,d] = softplus( x_dbl[m,0:32] @ dt_proj_w^T + dt_b ), 16x64 strip,
// K=32 -> single WMMA step per accumulator, epilogue fused.
// ---------------------------------------------------------------------------
__global__ void gemm_dt_softplus4(const float* __restrict__ xdbl,
                                  const float* __restrict__ Wdt,
                                  const float* __restrict__ bdt,
                                  float* __restrict__ delta) {
    const int wave = (int)((blockIdx.x * blockDim.x + threadIdx.x) >> 5);
    const int lane = threadIdx.x & 31;
    const int tm = wave >> 2, tg = wave & 3;         // ngroups = 256/64 = 4
    const int row0 = tm << 4, col0 = tg << 6;

    const float* arow = xdbl + (size_t)(row0 + (lane & 15)) * XDBLW;
    v16h a = load_afrag(arow, 0, lane);
    v8f acc[4];
#pragma unroll
    for (int j = 0; j < 4; ++j) {
        const float* wrow = Wdt + (size_t)(col0 + (j << 4) + (lane & 15)) * DTRANK;
        v16h b = load_bfrag(wrow, 0, lane);
        v8f c = {};
        acc[j] = __builtin_amdgcn_wmma_f32_16x16x32_f16(false, a, false, b,
                                                        (short)0, c, false, false);
    }
    const int mb = row0 + ((lane >> 4) << 3);
#pragma unroll
    for (int j = 0; j < 4; ++j) {
        const int n = col0 + (j << 4) + (lane & 15);
        const float bias = bdt[n];
#pragma unroll
        for (int i = 0; i < 8; ++i) {
            float v = acc[j][i] + bias;
            v = (v > 20.f) ? v : log1pf(__expf(v));  // softplus
            delta[(size_t)(mb + i) * D2 + n] = v;
        }
    }
}

// ---------------------------------------------------------------------------
// Depthwise conv (K=3, SAME, cross-correlation) + SiLU for both halves of xz.
// ch<256 -> xc[m, ch];  ch>=256 -> cat[m, ch] (right half = z branch)
// ---------------------------------------------------------------------------
__global__ void conv3_silu(const float* __restrict__ xz,
                           const float* __restrict__ wx, const float* __restrict__ bx,
                           const float* __restrict__ wz, const float* __restrict__ bz,
                           float* __restrict__ xc, float* __restrict__ cat) {
    const int idx = blockIdx.x * blockDim.x + threadIdx.x;   // BL*512 threads
    const int ch = idx & (DINNER - 1);
    const int m  = idx >> 9;
    const int l  = m & (SEQLEN - 1);

    const float xm1 = (l > 0)          ? xz[(size_t)(m - 1) * DINNER + ch] : 0.f;
    const float x0  =                    xz[(size_t)m       * DINNER + ch];
    const float xp1 = (l < SEQLEN - 1) ? xz[(size_t)(m + 1) * DINNER + ch] : 0.f;

    const int c = ch & (D2 - 1);
    const float* w  = (ch < D2) ? wx : wz;
    const float* bb = (ch < D2) ? bx : bz;
    float v = w[c * 3 + 0] * xm1 + w[c * 3 + 1] * x0 + w[c * 3 + 2] * xp1 + bb[c];
    v = v / (1.f + __expf(-v));                               // SiLU
    if (ch < D2) xc [(size_t)m * D2     + c]      = v;
    else         cat[(size_t)m * DINNER + D2 + c] = v;
}

// ---------------------------------------------------------------------------
// Chunked selective scan. Recurrence h <- a*h + v with a = exp(delta*A) is
// linear, so: pass1 computes per-chunk local end-state (h0=0) and total decay
// exp(A*sum(delta)); pass2 serially combines 32 chunk summaries per state;
// pass3 replays each chunk from its carry-in and writes y.
// Thread layout (pass 1/3): t = ((b*NCHUNK + c)*D2 + d)*16 + n  -> 16-lane
// groups share (b,c,d): broadcast scalar loads + shfl reduction stay in-group.
// ---------------------------------------------------------------------------
__global__ void scan_pass1(const float* __restrict__ xc,
                           const float* __restrict__ delta,
                           const float* __restrict__ xdbl,
                           const float* __restrict__ A_log,
                           float* __restrict__ hloc, float* __restrict__ asum) {
    const int t = blockIdx.x * blockDim.x + threadIdx.x;   // 262144 threads
    const int n = t & (NSTATE - 1);
    const int q = t >> 4;
    const int d = q & (D2 - 1);
    const int bc = q >> 8;                                 // 0..63
    const int c = bc & (NCHUNK - 1);
    const int b = bc >> 5;

    const float A = -__expf(A_log[d * NSTATE + n]);
    float h = 0.f, dsum = 0.f;
    const size_t m0 = (size_t)b * SEQLEN + (size_t)c * CLEN;
    for (int l = 0; l < CLEN; ++l) {
        const size_t m = m0 + l;
        const float dv = delta[m * D2 + d];
        const float u  = xc   [m * D2 + d];
        const float bn = xdbl [m * XDBLW + DTRANK + n];
        h = __expf(dv * A) * h + (dv * u) * bn;
        dsum += dv;
    }
    hloc[t] = h;
    asum[t] = A * dsum;   // log of total chunk decay (exact: prod exp = exp sum)
}

__global__ void scan_pass2(float* __restrict__ hloc,      // in: local, out: carry-in
                           const float* __restrict__ asum) {
    const int t = blockIdx.x * blockDim.x + threadIdx.x;  // 8192 threads (b,d,n)
    const int n = t & (NSTATE - 1);
    const int d = (t >> 4) & (D2 - 1);
    const int b = t >> 12;
    float carry = 0.f;
#pragma unroll 1
    for (int c = 0; c < NCHUNK; ++c) {
        const int idx = (((b * NCHUNK + c) * D2 + d) << 4) + n;
        const float hl = hloc[idx];
        const float as = asum[idx];
        hloc[idx] = carry;                 // carry-in for this chunk
        carry = __expf(as) * carry + hl;   // propagate across chunk
    }
}

__global__ void scan_pass3(const float* __restrict__ xc,
                           const float* __restrict__ delta,
                           const float* __restrict__ xdbl,
                           const float* __restrict__ A_log,
                           const float* __restrict__ Dp,
                           const float* __restrict__ hloc,  // carry-ins
                           float* __restrict__ cat) {
    const int t = blockIdx.x * blockDim.x + threadIdx.x;   // 262144 threads
    const int n = t & (NSTATE - 1);
    const int q = t >> 4;
    const int d = q & (D2 - 1);
    const int bc = q >> 8;
    const int c = bc & (NCHUNK - 1);
    const int b = bc >> 5;

    const float A  = -__expf(A_log[d * NSTATE + n]);
    const float Dv = Dp[d];
    float h = hloc[t];
    const size_t m0 = (size_t)b * SEQLEN + (size_t)c * CLEN;
    for (int l = 0; l < CLEN; ++l) {
        const size_t m = m0 + l;
        const float dv = delta[m * D2 + d];
        const float u  = xc   [m * D2 + d];
        const float bn = xdbl [m * XDBLW + DTRANK + n];
        const float cn = xdbl [m * XDBLW + DTRANK + NSTATE + n];
        h = __expf(dv * A) * h + (dv * u) * bn;
        float s = h * cn;                                  // 16-lane reduce
        s += __shfl_xor(s, 1, 32);
        s += __shfl_xor(s, 2, 32);
        s += __shfl_xor(s, 4, 32);
        s += __shfl_xor(s, 8, 32);
        if (n == 0) cat[m * DINNER + d] = s + Dv * u;
    }
}

// ---------------------------------------------------------------------------
extern "C" void kernel_launch(void* const* d_in, const int* in_sizes, int n_in,
                              void* d_out, int out_size, void* d_ws, size_t ws_size,
                              hipStream_t stream) {
    const float* H        = (const float*)d_in[0];   // (B, L, 512)
    const float* in_proj  = (const float*)d_in[1];   // (512, 512)
    const float* x_proj   = (const float*)d_in[2];   // (64, 256)
    const float* dt_proj  = (const float*)d_in[3];   // (256, 32)
    const float* dt_b     = (const float*)d_in[4];   // (256,)
    const float* A_log    = (const float*)d_in[5];   // (256, 16)
    const float* D_param  = (const float*)d_in[6];   // (256,)
    const float* conv_x_w = (const float*)d_in[7];   // (256,1,3)
    const float* conv_x_b = (const float*)d_in[8];   // (256,)
    const float* conv_z_w = (const float*)d_in[9];   // (256,1,3)
    const float* conv_z_b = (const float*)d_in[10];  // (256,)
    const float* out_proj = (const float*)d_in[11];  // (512, 512)
    float* out = (float*)d_out;                      // (B, L, 512) fp32

    // Workspace carve-up (fp32): ~102 MB total.
    float* xz    = (float*)d_ws;                     // (BL, 512)
    float* xc    = xz    + (size_t)BL * DINNER;      // (BL, 256)
    float* cat   = xc    + (size_t)BL * D2;          // (BL, 512) [y | zc]
    float* xdbl  = cat   + (size_t)BL * DINNER;      // (BL, 64)
    float* delta = xdbl  + (size_t)BL * XDBLW;       // (BL, 256)
    float* hloc  = delta + (size_t)BL * D2;          // (B*NCHUNK*D2*16) = 256K
    float* asum  = hloc  + (size_t)BATCH * NCHUNK * D2 * NSTATE;
    (void)ws_size; (void)in_sizes; (void)n_in; (void)out_size;

    const int SCAN_T = BATCH * NCHUNK * D2 * NSTATE;          // 262144

    // 1) xz = H @ in_proj^T : (1024 M-tiles) x (8 N-groups) waves
    gemm_f16wmma4<DMODEL, DINNER / 64><<<1024, 256, 0, stream>>>(
        H, in_proj, xz, DMODEL, DMODEL, DINNER);
    // 2) depthwise conv + SiLU (x -> xc, z -> cat right half)
    conv3_silu<<<(BL * DINNER) / 256, 256, 0, stream>>>(
        xz, conv_x_w, conv_x_b, conv_z_w, conv_z_b, xc, cat);
    // 3) x_dbl = xc @ x_proj^T : 1024 x 1 groups
    gemm_f16wmma4<D2, XDBLW / 64><<<128, 256, 0, stream>>>(
        xc, x_proj, xdbl, D2, D2, XDBLW);
    // 4) delta = softplus(x_dbl[:, :32] @ dt_proj^T + b) : 1024 x 4 groups
    gemm_dt_softplus4<<<512, 256, 0, stream>>>(xdbl, dt_proj, dt_b, delta);
    // 5) chunked selective scan -> cat left half
    scan_pass1<<<SCAN_T / 256, 256, 0, stream>>>(xc, delta, xdbl, A_log,
                                                 hloc, asum);
    scan_pass2<<<(BATCH * D2 * NSTATE) / 256, 256, 0, stream>>>(hloc, asum);
    scan_pass3<<<SCAN_T / 256, 256, 0, stream>>>(xc, delta, xdbl, A_log,
                                                 D_param, hloc, cat);
    // 6) out = cat @ out_proj^T
    gemm_f16wmma4<DINNER, DMODEL / 64><<<1024, 256, 0, stream>>>(
        cat, out_proj, out, DINNER, DINNER, DMODEL);
}
